// MonarchAttention_21543555956866
// MI455X (gfx1250) — matrix operand order repaired and checked by reading, not compiled, not run on hardware
//
#include <hip/hip_runtime.h>
#include <hip/hip_bf16.h>

// ---------------------------------------------------------------------------
// MonarchAttention for MI455X (gfx1250, wave32, WMMA).
//   B=2, S=2048, DIM=1024, H=16, D=64, B1=B2=8, F=32, NUM_ITERS=2
// Pipeline:
//   1) f32->f16 convert of x / wq / wk / wv into workspace
//   2) QKV projection GEMMs via v_wmma_f32_16x16x32_f16 (f32 accumulate)
//   3) in-place RMSNorm + RoPE + D^-0.25 scale on q,k (wave32 row reduce)
//   4) monarch iteration: one workgroup per (b,h,g), everything LDS-resident
// Workspace usage: 14 MB (f16 inputs) + 48 MB (q/k/v f32) = 62 MB.
// ---------------------------------------------------------------------------

#define BB      2
#define SS      2048
#define DIMC    1024
#define HH      16
#define DD      64
#define B1C     8
#define B2C     8
#define FF      32
#define NITERS  2
#define EPSF    1e-6f
#define SCALEF  0.35355339059327373f   // 64^(-1/4)

typedef __attribute__((ext_vector_type(16))) _Float16 v16h;
typedef __attribute__((ext_vector_type(8)))  _Float16 v8h;
typedef __attribute__((ext_vector_type(8)))  float    v8f;

// -------------------------------- convert ----------------------------------
__global__ void cvt_f32_f16(const float* __restrict__ in, _Float16* __restrict__ out, int n) {
    int i = blockIdx.x * blockDim.x + threadIdx.x;
    if (i < n) out[i] = (_Float16)in[i];
}

// ----------------------------- QKV WMMA GEMM -------------------------------
// C[row][col] = sum_k X[row][k] * W[col][k]   (X: 4096x1024 f16, W: 1024x1024 f16)
// 256 threads = 8 waves; wave computes one 16x16 tile; block covers 64x32.
__global__ void qkv_gemm(const _Float16* __restrict__ X,
                         const _Float16* __restrict__ Wq,
                         const _Float16* __restrict__ Wk,
                         const _Float16* __restrict__ Wv,
                         float* __restrict__ Q,
                         float* __restrict__ K,
                         float* __restrict__ V) {
    const int which = blockIdx.z;
    const _Float16* __restrict__ W = (which == 0) ? Wq : (which == 1) ? Wk : Wv;
    float* __restrict__ Out        = (which == 0) ? Q  : (which == 1) ? K  : V;

    const int wave = threadIdx.x >> 5;
    const int lane = threadIdx.x & 31;
    const int m    = lane & 15;
    const int hi   = lane >> 4;

    const int row0 = blockIdx.y * 64 + (wave >> 1) * 16;
    const int col0 = blockIdx.x * 32 + (wave & 1) * 16;

    const _Float16* __restrict__ arow = X + (size_t)(row0 + m) * DIMC;
    const _Float16* __restrict__ brow = W + (size_t)(col0 + m) * DIMC;

    v8f c = {};
    for (int kk = 0; kk < DIMC; kk += 32) {
        __builtin_prefetch(arow + kk + 128, 0, 0);   // global_prefetch_b8
        __builtin_prefetch(brow + kk + 128, 0, 0);
        // A fragment (16x32 f16): lane hi selects K interleave {hi*8..+7, 16+hi*8..+7}
        v8h a0 = *(const v8h*)(arow + kk + hi * 8);
        v8h a1 = *(const v8h*)(arow + kk + 16 + hi * 8);
        // B fragment (32x16 f16): lane hi selects K interleave {hi*8..+7, 16+hi*8..+7}
        v8h b0 = *(const v8h*)(brow + kk + hi * 8);
        v8h b1 = *(const v8h*)(brow + kk + 16 + hi * 8);
        v16h a, b;
#pragma unroll
        for (int e = 0; e < 8; ++e) { a[e] = a0[e]; a[e + 8] = a1[e]; b[e] = b0[e]; b[e + 8] = b1[e]; }
        c = __builtin_amdgcn_wmma_f32_16x16x32_f16(false, a, false, b, (short)0, c, false, false);
    }
    // D layout: VGPR r -> row r + 8*hi, col = lane&15
#pragma unroll
    for (int r = 0; r < 8; ++r)
        Out[(size_t)(row0 + r + 8 * hi) * DIMC + (col0 + m)] = c[r];
}

// ------------------------- RMSNorm + RoPE + scale --------------------------
// One wave32 per (b,s,h) row of 64; thread lane handles the rotary pair (2m,2m+1).
__global__ void rmsnorm_rope(float* __restrict__ t,
                             const float* __restrict__ gvec,
                             const float* __restrict__ cosb,
                             const float* __restrict__ sinb) {
    const int ridx = blockIdx.x * 8 + (threadIdx.x >> 5);   // over B*S*H
    const int lane = threadIdx.x & 31;
    const int h  = ridx % HH;
    const int bs = ridx / HH;
    const int s  = bs % SS;

    float* __restrict__ row = t + (size_t)bs * DIMC + h * DD;
    float x0 = row[2 * lane];
    float x1 = row[2 * lane + 1];

    float ssq = x0 * x0 + x1 * x1;
#pragma unroll
    for (int off = 16; off > 0; off >>= 1) ssq += __shfl_xor(ssq, off, 32);
    const float inv = rsqrtf(ssq * (1.0f / 64.0f) + EPSF);

    const float g0 = gvec[2 * lane], g1 = gvec[2 * lane + 1];
    const float c0 = cosb[s * DD + 2 * lane], c1 = cosb[s * DD + 2 * lane + 1];
    const float s0 = sinb[s * DD + 2 * lane], s1 = sinb[s * DD + 2 * lane + 1];

    const float n0 = x0 * inv * g0;
    const float n1 = x1 * inv * g1;
    row[2 * lane]     = (n0 * c0 - n1 * s0) * SCALEF;
    row[2 * lane + 1] = (n1 * c1 + n0 * s1) * SCALEF;
}

// ------------------------------ monarch core -------------------------------
// One workgroup per (b,h,g); 256 threads; ~162.5 KB dynamic LDS (WGP has 320KB).
// LDS: qs[i][j][d] | L[f][j][k][i] | R[f][k][j][l] | tmp[.][.][d] | bR | cR | cL
__global__ void monarch_core(const float* __restrict__ qn,
                             const float* __restrict__ kn,
                             const float* __restrict__ vn,
                             float* __restrict__ out) {
    extern __shared__ float sm[];
    float* qs  = sm;             // 4096:  (i*8+j)*64+d
    float* Lb  = qs  + 4096;     // 16384: f*512 + j*64 + k*8 + i
    float* Rb  = Lb  + 16384;    // 16384: f*512 + k*64 + j*8 + l
    float* tmp = Rb  + 16384;    // 4096
    float* bRb = tmp + 4096;     // 512:   k*64 + j*8 + l
    float* cRb = bRb + 512;      // 64:    k*8 + j
    float* cLb = cRb + 64;       // 64:    j*8 + k

    const int tid = threadIdx.x;
    const int g = blockIdx.x, h = blockIdx.y, b = blockIdx.z;

    const float* __restrict__ qbase = qn + ((size_t)b * SS) * DIMC + h * DD;
    const float* __restrict__ kbase = kn + ((size_t)b * SS) * DIMC + h * DD;
    const float* __restrict__ vbase = vn + ((size_t)b * SS) * DIMC + h * DD;

    // load q tile for block g: s = (g*8+i)*8+j
    for (int idx = tid; idx < 4096; idx += 256) {
        int d = idx & 63, ij = idx >> 6;
        int i = ij >> 3, j = ij & 7;
        int s = (g * 8 + i) * 8 + j;
        qs[idx] = qbase[(size_t)s * DIMC + d];
    }
    // init L[f][j][k][i] = (k==i)  (only f<=g ever used; mask handled by loop bounds)
    for (int idx = tid; idx < 16384; idx += 256) {
        int i = idx & 7, k = (idx >> 3) & 7;
        Lb[idx] = (i == k) ? 1.0f : 0.0f;
    }
    __syncthreads();

    for (int iter = 0; iter < NITERS; ++iter) {
        for (int f = 0; f <= g; ++f) {
            const float* __restrict__ kf = kbase + (size_t)(f * 64) * DIMC; // rows (k*8+l)
            float* Lf = Lb + f * 512;
            float* Rf = Rb + f * 512;

            // aR[k][j][d] = sum_i L[f][j][k][i] * qs[i][j][d]
            for (int idx = tid; idx < 4096; idx += 256) {
                int d = idx & 63, kj = idx >> 6;
                int k = kj >> 3, j = kj & 7;
                float acc = 0.f;
#pragma unroll
                for (int i = 0; i < 8; ++i) acc += Lf[j * 64 + k * 8 + i] * qs[(i * 8 + j) * 64 + d];
                tmp[idx] = acc;
            }
            __syncthreads();

            // bR[k][j][l] = sum_d aR[k][j][d] * k_f[k][l][d] ; cR[k][j] = sum_i L
            for (int idx = tid; idx < 512; idx += 256) {
                int l = idx & 7, kj = idx >> 3;
                int j = kj & 7, k = kj >> 3;
                const float* kr = kf + (size_t)(k * 8 + l) * DIMC;
                const float* ar = tmp + (k * 8 + j) * 64;
                float acc = 0.f;
                for (int d = 0; d < 64; ++d) acc += ar[d] * kr[d];
                bRb[idx] = acc;
            }
            if (tid < 64) {
                int j = tid & 7, k = tid >> 3;
                float acc = 0.f;
#pragma unroll
                for (int i = 0; i < 8; ++i) acc += Lf[j * 64 + k * 8 + i];
                cRb[tid] = acc;
            }
            __syncthreads();

            // R[f][k][j][l] = softmax_l(bR/(cR+eps)); cL[j][k] = sum_l R*logR
            if (tid < 64) {
                int j = tid & 7, k = tid >> 3;
                float inv = 1.0f / (cRb[k * 8 + j] + EPSF);
                float vals[8], mx = -1e30f;
#pragma unroll
                for (int l = 0; l < 8; ++l) { vals[l] = bRb[k * 64 + j * 8 + l] * inv; mx = fmaxf(mx, vals[l]); }
                float sum = 0.f;
#pragma unroll
                for (int l = 0; l < 8; ++l) { vals[l] = __expf(vals[l] - mx); sum += vals[l]; }
                float rinv = 1.0f / sum, ent = 0.f;
#pragma unroll
                for (int l = 0; l < 8; ++l) {
                    float r = vals[l] * rinv;
                    Rf[k * 64 + j * 8 + l] = r;
                    ent += (r > 0.f) ? r * __logf(r) : 0.f;
                }
                cLb[j * 8 + k] = ent;
            }
            __syncthreads();

            // aL[j][k][d] = sum_l R[f][k][j][l] * k_f[k][l][d]
            for (int idx = tid; idx < 4096; idx += 256) {
                int d = idx & 63, jk = idx >> 6;
                int k = jk & 7, j = jk >> 3;
                float acc = 0.f;
#pragma unroll
                for (int l = 0; l < 8; ++l) acc += Rf[k * 64 + j * 8 + l] * kf[(size_t)(k * 8 + l) * DIMC + d];
                tmp[idx] = acc;
            }
            __syncthreads();

            // M[f][j][k][i] = sum_d aL[j][k][d]*qs[i][j][d] - cL[j][k]  (overwrite L storage)
            for (int idx = tid; idx < 512; idx += 256) {
                int i = idx & 7, jk = idx >> 3;
                int k = jk & 7, j = jk >> 3;
                const float* al = tmp + (j * 8 + k) * 64;
                const float* qr = qs + (i * 8 + j) * 64;
                float acc = 0.f;
                for (int d = 0; d < 64; ++d) acc += al[d] * qr[d];
                Lf[j * 64 + k * 8 + i] = acc - cLb[j * 8 + k];
            }
            __syncthreads();
        }

        // L[f][j][k][i] = softmax over (f<=g, k) of M, per (j,i); f>g stays masked out
        if (tid < 64) {
            int i = tid & 7, j = tid >> 3;
            float mx = -1e30f;
            for (int f = 0; f <= g; ++f)
#pragma unroll
                for (int k = 0; k < 8; ++k) mx = fmaxf(mx, Lb[f * 512 + j * 64 + k * 8 + i]);
            float sum = 0.f;
            for (int f = 0; f <= g; ++f)
#pragma unroll
                for (int k = 0; k < 8; ++k) {
                    float e = __expf(Lb[f * 512 + j * 64 + k * 8 + i] - mx);
                    Lb[f * 512 + j * 64 + k * 8 + i] = e;
                    sum += e;
                }
            float inv = 1.0f / sum;
            for (int f = 0; f <= g; ++f)
#pragma unroll
                for (int k = 0; k < 8; ++k) Lb[f * 512 + j * 64 + k * 8 + i] *= inv;
        }
        __syncthreads();
    }

    // out[i][j][d] = sum_{f<=g,k} L[f][j][k][i] * (sum_l R[f][k][j][l] * v_f[k][l][d])
    const int oij = tid >> 2;
    const int oi = oij >> 3, oj = oij & 7;
    const int dp = (tid & 3) * 16;
    float acc[16];
#pragma unroll
    for (int t = 0; t < 16; ++t) acc[t] = 0.f;

    for (int f = 0; f <= g; ++f) {
        const float* __restrict__ vf = vbase + (size_t)(f * 64) * DIMC;
        const float* Rf = Rb + f * 512;
        // o[j][k][d] into tmp
        for (int idx = tid; idx < 4096; idx += 256) {
            int d = idx & 63, jk = idx >> 6;
            int k = jk & 7, j = jk >> 3;
            float a = 0.f;
#pragma unroll
            for (int l = 0; l < 8; ++l) a += Rf[k * 64 + j * 8 + l] * vf[(size_t)(k * 8 + l) * DIMC + d];
            tmp[idx] = a;
        }
        __syncthreads();
#pragma unroll
        for (int k = 0; k < 8; ++k) {
            float lw = Lb[f * 512 + oj * 64 + k * 8 + oi];
            const float* orow = tmp + (oj * 8 + k) * 64 + dp;
#pragma unroll
            for (int t = 0; t < 16; ++t) acc[t] += lw * orow[t];
        }
        __syncthreads();
    }

    {
        int s = (g * 8 + oi) * 8 + oj;
        float* orow = out + ((size_t)b * SS + s) * DIMC + h * DD + dp;
#pragma unroll
        for (int t = 0; t < 16; ++t) orow[t] = acc[t];
    }
}

// ------------------------------- launcher ----------------------------------
extern "C" void kernel_launch(void* const* d_in, const int* in_sizes, int n_in,
                              void* d_out, int out_size, void* d_ws, size_t ws_size,
                              hipStream_t stream) {
    (void)in_sizes; (void)n_in; (void)out_size; (void)ws_size;

    const float* x    = (const float*)d_in[0];
    const float* cosb = (const float*)d_in[1];
    const float* sinb = (const float*)d_in[2];
    const float* wq   = (const float*)d_in[3];
    const float* wk   = (const float*)d_in[4];
    const float* wv   = (const float*)d_in[5];
    const float* gq   = (const float*)d_in[6];
    const float* gk   = (const float*)d_in[7];
    float* out = (float*)d_out;

    const size_t M   = (size_t)BB * SS;      // 4096
    const size_t XN  = M * DIMC;             // 4.19M elems
    const size_t WN  = (size_t)DIMC * DIMC;  // 1.05M elems

    _Float16* xh  = (_Float16*)d_ws;
    _Float16* wqh = xh  + XN;
    _Float16* wkh = wqh + WN;
    _Float16* wvh = wkh + WN;
    float* qraw = (float*)(wvh + WN);
    float* kraw = qraw + XN;
    float* vraw = kraw + XN;

    cvt_f32_f16<<<(int)(XN / 256), 256, 0, stream>>>(x,  xh,  (int)XN);
    cvt_f32_f16<<<(int)(WN / 256), 256, 0, stream>>>(wq, wqh, (int)WN);
    cvt_f32_f16<<<(int)(WN / 256), 256, 0, stream>>>(wk, wkh, (int)WN);
    cvt_f32_f16<<<(int)(WN / 256), 256, 0, stream>>>(wv, wvh, (int)WN);

    dim3 ggrid(DIMC / 32, (unsigned)(M / 64), 3);
    qkv_gemm<<<ggrid, 256, 0, stream>>>(xh, wqh, wkh, wvh, qraw, kraw, vraw);

    const int nrows = BB * SS * HH;          // 65536
    rmsnorm_rope<<<nrows / 8, 256, 0, stream>>>(qraw, gq, cosb, sinb);
    rmsnorm_rope<<<nrows / 8, 256, 0, stream>>>(kraw, gk, cosb, sinb);

    const int smem_bytes = 41600 * sizeof(float);  // 166,400 B (<320KB/WGP)
    (void)hipFuncSetAttribute((const void*)monarch_core,
                              hipFuncAttributeMaxDynamicSharedMemorySize, smem_bytes);
    dim3 mgrid(FF, HH, BB);                  // (32,16,2) = 1024 workgroups
    monarch_core<<<mgrid, 256, smem_bytes, stream>>>(qraw, kraw, vraw, out);
}
